// EdgeAwareAttention_52012053954934
// MI455X (gfx1250) — matrix operand order repaired and checked by reading, not compiled, not run on hardware
//
#include <hip/hip_runtime.h>
#include <hip/hip_bf16.h>

typedef __bf16 bf16_t;
typedef __attribute__((ext_vector_type(16))) __bf16 v16bf;
typedef __attribute__((ext_vector_type(8)))  __bf16 v8bf;
typedef __attribute__((ext_vector_type(8)))  float  v8f;

#define LN_EPS 1e-5f

// ---------------- float <-> ordered-uint encoding for atomicMax ----------------
__device__ __forceinline__ unsigned encf(float f) {
    unsigned u = __float_as_uint(f);
    return (u & 0x80000000u) ? ~u : (u | 0x80000000u);
}
__device__ __forceinline__ float decf(unsigned u) {
    return (u & 0x80000000u) ? __uint_as_float(u & 0x7fffffffu) : __uint_as_float(~u);
}
#define MAXENC_NEG_INF 0x007FFFFFu   // encf(-inf)

// ---------------- conversion kernels ----------------
__global__ void conv_w_kernel(const float* __restrict__ Wq, const float* __restrict__ Wk,
                              const float* __restrict__ Wv, const float* __restrict__ Wo,
                              bf16_t* __restrict__ wbf) {
    int i = blockIdx.x * 256 + threadIdx.x;
    if (i >= 4 * 16384) return;
    const float* srcs[4] = {Wq, Wk, Wv, Wo};
    wbf[i] = (bf16_t)srcs[i >> 14][i & 16383];
}

__global__ void conv_x_kernel(const float* __restrict__ x, bf16_t* __restrict__ xbf,
                              long long total) {
    long long i = (long long)blockIdx.x * 256 + threadIdx.x;
    if (i >= total) return;
    xbf[i] = (bf16_t)x[i];
}

// ---------------- WMMA GEMM tile: C[16x16] += A[16xK=128] * W^T ----------------
// A: M x 128 row-major bf16.  W: 128 x 128 row-major bf16 (we compute A @ W^T,
// so B(KxN) column n == W row n, contiguous in K).  C: M x 128 row-major f32.
__device__ __forceinline__ void wmma_tile_128(const bf16_t* __restrict__ A,
                                              const bf16_t* __restrict__ W,
                                              float* __restrict__ C,
                                              int rowBase, int colBase) {
    int lane = threadIdx.x & 31;
    int m    = lane & 15;        // row within tile (A) / col within tile (B)
    int hi   = lane >> 4;        // lane-half selector
    v8f c = {0.f, 0.f, 0.f, 0.f, 0.f, 0.f, 0.f, 0.f};

    const bf16_t* arow = A + (size_t)(rowBase + m) * 128;
    const bf16_t* brow = W + (size_t)(colBase + m) * 128;

#pragma unroll
    for (int kc = 0; kc < 4; ++kc) {
        int kb = kc * 32;
        // A 16x32 bf16, contiguous k order used here (same as B): lanes 0-15: K 0..15; lanes 16-31: K 16..31
        v16bf a = *(const v16bf*)(arow + kb + hi * 16);
        // B 32x16 bf16, contiguous k order used here (same as A): lanes 0-15 hold K=0..15 of col n; lanes 16-31 hold K=16..31
        v16bf b = *(const v16bf*)(brow + kb + hi * 16);
        c = __builtin_amdgcn_wmma_f32_16x16x32_bf16(false, a, false, b, (short)0, c,
                                                    false, false);
    }
    // D layout: VGPR v -> row = v + hi*8, col = lane&15
    float* cp = C + (size_t)rowBase * 128 + colBase + m;
#pragma unroll
    for (int v = 0; v < 8; ++v) cp[(size_t)(v + hi * 8) * 128] = c[v];
}

// QKV: grid (N/16, 3); block 256 = 8 waves, wave w -> column tile w
__global__ void qkv_gemm_kernel(const bf16_t* __restrict__ xbf,
                                const bf16_t* __restrict__ wbf,
                                float* __restrict__ qkv, int N) {
    int sel = blockIdx.y;                       // 0=Q,1=K,2=V
    int rowBase = blockIdx.x * 16;
    int colBase = (threadIdx.x >> 5) * 16;
    if (rowBase >= N) return;
    wmma_tile_128(xbf, wbf + (size_t)sel * 16384,
                  qkv + (size_t)sel * N * 128, rowBase, colBase);
}

// Output projection: grid (N/16); block 256
__global__ void out_gemm_kernel(const bf16_t* __restrict__ aggbf,
                                const bf16_t* __restrict__ wo_bf,
                                float* __restrict__ outraw, int N) {
    int rowBase = blockIdx.x * 16;
    int colBase = (threadIdx.x >> 5) * 16;
    if (rowBase >= N) return;
    wmma_tile_128(aggbf, wo_bf, outraw, rowBase, colBase);
}

// ---------------- init kernels ----------------
__global__ void init_md_kernel(unsigned* __restrict__ maxenc, float* __restrict__ denom,
                               int n8) {
    int i = blockIdx.x * 256 + threadIdx.x;
    if (i >= n8) return;
    maxenc[i] = MAXENC_NEG_INF;
    denom[i]  = 0.f;
}
__global__ void zero_f_kernel(float* __restrict__ p, long long total) {
    long long i = (long long)blockIdx.x * 256 + threadIdx.x;
    if (i < total) p[i] = 0.f;
}

// ---------------- edge scores + segment max (one wave per edge) ----------------
__global__ void edge_scores_kernel(const float* __restrict__ Q, const float* __restrict__ K,
                                   const long long* __restrict__ ei,
                                   const float* __restrict__ ew,
                                   const float* __restrict__ We,
                                   float* __restrict__ scores,
                                   unsigned* __restrict__ maxenc, int E) {
    int wave = threadIdx.x >> 5, lane = threadIdx.x & 31;
    long long e = (long long)blockIdx.x * 8 + wave;
    if (e >= E) return;
    long long src = ei[e];
    long long dst = ei[(long long)E + e];
    float4 q = ((const float4*)(Q + src * 128))[lane];
    float4 k = ((const float4*)(K + dst * 128))[lane];
    float p = q.x * k.x + q.y * k.y + q.z * k.z + q.w * k.w;
    // reduce within each 4-lane head group (head h = lanes 4h..4h+3 cover dims 16h..16h+15)
    p += __shfl_xor(p, 1, 32);
    p += __shfl_xor(p, 2, 32);
    if ((lane & 3) == 0) {
        int h = lane >> 2;
        float s = p * 0.25f + ew[e] * We[h];   // scale = 1/sqrt(16)
        scores[e * 8 + h] = s;
        atomicMax(maxenc + src * 8 + h, encf(s));
    }
}

// ---------------- fused exp + denom + weighted-V aggregation ----------------
__global__ void edge_agg_kernel(const float* __restrict__ V,
                                const float* __restrict__ scores,
                                const unsigned* __restrict__ maxenc,
                                const long long* __restrict__ ei,
                                float* __restrict__ denom,
                                float* __restrict__ agg, int E) {
    int wave = threadIdx.x >> 5, lane = threadIdx.x & 31;
    long long e = (long long)blockIdx.x * 8 + wave;
    if (e >= E) return;
    long long src = ei[e];
    long long dst = ei[(long long)E + e];
    int h = lane >> 2;
    float s  = scores[e * 8 + h];
    float m  = decf(maxenc[src * 8 + h]);
    float eh = __expf(s - m);
    if ((lane & 3) == 0) atomicAdd(denom + src * 8 + h, eh);
    float4 v = ((const float4*)(V + dst * 128))[lane];
    float* ag = agg + src * 128 + lane * 4;
    atomicAdd(ag + 0, eh * v.x);
    atomicAdd(ag + 1, eh * v.y);
    atomicAdd(ag + 2, eh * v.z);
    atomicAdd(ag + 3, eh * v.w);
}

// ---------------- normalize by denom, convert to bf16 ----------------
__global__ void norm_bf16_kernel(const float* __restrict__ agg,
                                 const float* __restrict__ denom,
                                 bf16_t* __restrict__ aggbf, long long total) {
    long long i = (long long)blockIdx.x * 256 + threadIdx.x;
    if (i >= total) return;
    long long n = i >> 7;
    int d = (int)(i & 127);
    float dn = denom[n * 8 + (d >> 4)];
    float v = (dn > 0.f) ? agg[i] / dn : 0.f;
    aggbf[i] = (bf16_t)v;
}

// ---------------- residual + LayerNorm (one wave per row) ----------------
__global__ void resid_ln_kernel(const float* __restrict__ x,
                                const float* __restrict__ outraw,
                                const float* __restrict__ gamma,
                                const float* __restrict__ beta,
                                float* __restrict__ y, int N) {
    int wave = threadIdx.x >> 5, lane = threadIdx.x & 31;
    long long row = (long long)blockIdx.x * 8 + wave;
    if (row >= N) return;
    float4 o  = ((const float4*)(outraw + row * 128))[lane];
    float4 xi = ((const float4*)(x + row * 128))[lane];
    float4 yv;
    yv.x = xi.x + o.x; yv.y = xi.y + o.y; yv.z = xi.z + o.z; yv.w = xi.w + o.w;
    float s  = yv.x + yv.y + yv.z + yv.w;
    float s2 = yv.x * yv.x + yv.y * yv.y + yv.z * yv.z + yv.w * yv.w;
#pragma unroll
    for (int off = 1; off < 32; off <<= 1) {
        s  += __shfl_xor(s,  off, 32);
        s2 += __shfl_xor(s2, off, 32);
    }
    float mu  = s * (1.f / 128.f);
    float var = s2 * (1.f / 128.f) - mu * mu;
    float inv = rsqrtf(var + LN_EPS);
    float4 g  = ((const float4*)gamma)[lane];
    float4 bb = ((const float4*)beta)[lane];
    float4 r;
    r.x = (yv.x - mu) * inv * g.x + bb.x;
    r.y = (yv.y - mu) * inv * g.y + bb.y;
    r.z = (yv.z - mu) * inv * g.z + bb.z;
    r.w = (yv.w - mu) * inv * g.w + bb.w;
    ((float4*)(y + row * 128))[lane] = r;
}

// ---------------- host launch ----------------
extern "C" void kernel_launch(void* const* d_in, const int* in_sizes, int n_in,
                              void* d_out, int out_size, void* d_ws, size_t ws_size,
                              hipStream_t stream) {
    (void)n_in; (void)out_size; (void)ws_size;
    const float*     x     = (const float*)d_in[0];
    const long long* ei    = (const long long*)d_in[1];
    const float*     ew    = (const float*)d_in[2];
    const float*     Wq    = (const float*)d_in[3];
    const float*     Wk    = (const float*)d_in[4];
    const float*     Wv    = (const float*)d_in[5];
    const float*     We    = (const float*)d_in[6];
    const float*     Wo    = (const float*)d_in[7];
    const float*     gamma = (const float*)d_in[8];
    const float*     beta  = (const float*)d_in[9];

    const int N = in_sizes[0] / 128;       // 50000
    const int E = in_sizes[2];             // 800000
    const long long N128 = (long long)N * 128;

    // ---- workspace layout (256B aligned); Q/agg, K/aggbf, scores/outraw reuse ----
    char* ws = (char*)d_ws;
    auto al = [](size_t o) { return (o + 255) & ~(size_t)255; };
    size_t off = 0;
    bf16_t* xbf = (bf16_t*)(ws + off);  off = al(off + (size_t)N128 * 2);
    bf16_t* wbf = (bf16_t*)(ws + off);  off = al(off + (size_t)4 * 16384 * 2);
    float* Qb   = (float*)(ws + off);   off = al(off + (size_t)N128 * 4);
    float* Kb   = (float*)(ws + off);   off = al(off + (size_t)N128 * 4);
    float* Vb   = (float*)(ws + off);   off = al(off + (size_t)N128 * 4);
    size_t scsz = (size_t)E * 8 * 4;  if (scsz < (size_t)N128 * 4) scsz = (size_t)N128 * 4;
    float* scores = (float*)(ws + off); off = al(off + scsz);
    unsigned* maxenc = (unsigned*)(ws + off); off = al(off + (size_t)N * 8 * 4);
    float* denom     = (float*)(ws + off);    off = al(off + (size_t)N * 8 * 4);
    float*  aggf   = Qb;              // Q dead after edge_scores
    bf16_t* aggbf  = (bf16_t*)Kb;     // K dead after edge_scores
    float*  outraw = scores;          // scores dead after edge_agg

    // 1) weights + x -> bf16
    conv_w_kernel<<<(4 * 16384 + 255) / 256, 256, 0, stream>>>(Wq, Wk, Wv, Wo, wbf);
    conv_x_kernel<<<(int)((N128 + 255) / 256), 256, 0, stream>>>(x, xbf, N128);
    // 2) Q,K,V via WMMA bf16
    dim3 gq(N / 16, 3);
    qkv_gemm_kernel<<<gq, 256, 0, stream>>>(xbf, wbf, Qb, N);
    // 3) init segment-max / denom
    init_md_kernel<<<(N * 8 + 255) / 256, 256, 0, stream>>>(maxenc, denom, N * 8);
    // 4) per-edge scores + atomic segment max
    edge_scores_kernel<<<(E + 7) / 8, 256, 0, stream>>>(Qb, Kb, ei, ew, We, scores,
                                                        maxenc, E);
    // 5) zero agg (reuses Q region, so must follow edge_scores)
    zero_f_kernel<<<(int)((N128 + 255) / 256), 256, 0, stream>>>(aggf, N128);
    // 6) fused exp + denom + weighted V scatter
    edge_agg_kernel<<<(E + 7) / 8, 256, 0, stream>>>(Vb, scores, maxenc, ei, denom,
                                                     aggf, E);
    // 7) normalize + bf16
    norm_bf16_kernel<<<(int)((N128 + 255) / 256), 256, 0, stream>>>(aggf, denom, aggbf,
                                                                    N128);
    // 8) output projection via WMMA bf16
    out_gemm_kernel<<<N / 16, 256, 0, stream>>>(aggbf, wbf + (size_t)3 * 16384, outraw, N);
    // 9) residual + LayerNorm
    resid_ln_kernel<<<(N + 7) / 8, 256, 0, stream>>>(x, outraw, gamma, beta,
                                                     (float*)d_out, N);
}
